// GraphNodeEdgeAction_2697239462645
// MI455X (gfx1250) — matrix operand order repaired and checked by reading, not compiled, not run on hardware
//
#include <hip/hip_runtime.h>
#include <hip/hip_bf16.h>

// ---------------------------------------------------------------------------
// GIN forward for MI455X (gfx1250, wave32).  All GEMMs run on
// v_wmma_f32_16x16x32_bf16 with f32 accumulation; LayerNorm/ReLU/bias/residual
// are fused into the GEMM epilogue through LDS.  Edge aggregation uses
// global_atomic_add_f32 (L2-resident, 192MB L2 >> 64MB activations).
// ---------------------------------------------------------------------------

#define HID    256
#define NNODE  65536
#define BGR    128
#define NPER   512
#define EDGES  (NNODE * 32)
#define LNEPS  1e-5f

typedef __bf16 bf16;
typedef __attribute__((ext_vector_type(16))) __bf16 v16bf;
typedef __attribute__((ext_vector_type(8)))  __bf16 v8bf;
typedef __attribute__((ext_vector_type(8)))  float  v8f;

union Frag16 { v16bf v; v8bf h[2]; };
union FragC  { v8f v; float e[8]; };

__device__ __forceinline__ v8f wmma_bf16(v16bf a, v16bf b, v8f c) {
  // (neg_a, A, neg_b, B, c_mod, C, reuse_a, reuse_b)
  return __builtin_amdgcn_wmma_f32_16x16x32_bf16(false, a, false, b, (short)0, c,
                                                 false, false);
}

// --- transpose+convert weight: W[k][n] f32 -> Wt[n][k] bf16 -----------------
__global__ void k_wtrans(const float* __restrict__ W, bf16* __restrict__ Wt) {
  int n = blockIdx.x, k = threadIdx.x;
  Wt[n * HID + k] = (bf16)W[k * HID + n];
}

// --- embedding gather -------------------------------------------------------
__global__ void k_embed(const int* __restrict__ cls, const float* __restrict__ emb,
                        float* __restrict__ x) {
  int r = blockIdx.x, c = threadIdx.x;
  x[(size_t)r * HID + c] = emb[cls[r] * HID + c];
}

__global__ void k_copy(const float* __restrict__ a, float* __restrict__ b) {
  size_t i = (size_t)blockIdx.x * blockDim.x + threadIdx.x;
  b[i] = a[i];
}

__global__ void k_tobf(const float* __restrict__ a, bf16* __restrict__ b) {
  size_t i = (size_t)blockIdx.x * blockDim.x + threadIdx.x;
  b[i] = (bf16)a[i];
}

// --- agg[dst] += x[src] (one block per edge, one lane per feature) ----------
__global__ void k_scatter(const int* __restrict__ ei, const float* __restrict__ x,
                          float* __restrict__ agg) {
  int e = blockIdx.x, c = threadIdx.x;
  int s = ei[e], d = ei[EDGES + e];
  atomicAdd(&agg[(size_t)d * HID + c], x[(size_t)s * HID + c]);
}

// --- fused GEMM: out = post( A[Mx256](bf16) @ Wt^T + bias [,+resid][,LN][,ReLU])
// flags: bit0 = LayerNorm, bit1 = ReLU.  Block = 256 thr (8 waves), 64 rows x 256 cols.
__global__ __launch_bounds__(256)
void k_gemm(const bf16* __restrict__ A, const bf16* __restrict__ Wt,
            const float* __restrict__ bias,
            const float* __restrict__ lng, const float* __restrict__ lnb,
            const float* __restrict__ resid,
            float* __restrict__ outF, bf16* __restrict__ outBF,
            int M, int flags) {
  __shared__ float smem[64][HID];   // 64KB: tile staged for LN epilogue
  const int tid  = threadIdx.x;
  const int w    = tid >> 5, L = tid & 31;
  const int half = L >> 4, ln16 = L & 15;
  const int rowBlk = blockIdx.x * 64;
  const int wr = (w & 3) * 16;          // wave row offset inside block tile
  const int wc = (w >> 2) * 128;        // wave col offset (two col halves)
  const int growA = rowBlk + wr + ln16; // row this lane streams for the A frag

  FragC acc[8];
#pragma unroll
  for (int nt = 0; nt < 8; ++nt)
#pragma unroll
    for (int e = 0; e < 8; ++e) acc[nt].e[e] = 0.f;

  for (int kt = 0; kt < 8; ++kt) {      // K = 8 * 32 = 256
    Frag16 a;
    const bf16* ap = A + (size_t)growA * HID + kt * 32 + half * 8;
    a.h[0] = *(const v8bf*)(ap);
    a.h[1] = *(const v8bf*)(ap + 16);
#pragma unroll
    for (int nt = 0; nt < 8; ++nt) {
      int n = wc + nt * 16 + ln16;
      const bf16* bp = Wt + (size_t)n * HID + kt * 32 + half * 8;
      Frag16 b;
      b.h[0] = *(const v8bf*)(bp);
      b.h[1] = *(const v8bf*)(bp + 16);
      acc[nt].v = wmma_bf16(a.v, b.v, acc[nt].v);
    }
  }

  // stage C tile (+bias, +residual) into LDS
#pragma unroll
  for (int nt = 0; nt < 8; ++nt) {
    int lc = wc + nt * 16 + ln16;
#pragma unroll
    for (int v = 0; v < 8; ++v) {
      int lr = wr + v + half * 8;
      float val = acc[nt].e[v] + bias[lc];
      if (resid) val += resid[(size_t)(rowBlk + lr) * HID + lc];
      smem[lr][lc] = val;
    }
  }
  __syncthreads();

  if (flags & 1) {  // LayerNorm (4 threads per row, shfl reduce, rotated banks)
    int r = tid >> 2, q = tid & 3;
    float s = 0.f, ss = 0.f;
    for (int i = 0; i < 64; ++i) {
      int c = q * 64 + ((i + 4 * r) & 63);
      float v = smem[r][c];
      s += v; ss += v * v;
    }
    s  += __shfl_xor(s, 1);  s  += __shfl_xor(s, 2);
    ss += __shfl_xor(ss, 1); ss += __shfl_xor(ss, 2);
    float mu  = s * (1.f / HID);
    float var = ss * (1.f / HID) - mu * mu;
    float ri  = rsqrtf(var + LNEPS);
    size_t grow = (size_t)rowBlk + r;
    for (int i = 0; i < 64; ++i) {
      int c = q * 64 + ((i + 4 * r) & 63);
      float v = (smem[r][c] - mu) * ri * lng[c] + lnb[c];
      if (flags & 2) v = fmaxf(v, 0.f);
      if (outF)  outF [grow * HID + c] = v;
      if (outBF) outBF[grow * HID + c] = (bf16)v;
    }
  } else {
    for (int r = 0; r < 64; ++r) {
      float v = smem[r][tid];
      if (flags & 2) v = fmaxf(v, 0.f);
      size_t grow = (size_t)rowBlk + r;
      if (outF)  outF [grow * HID + tid] = v;
      if (outBF) outBF[grow * HID + tid] = (bf16)v;
    }
  }
}

// --- per-graph mean over 512 rows ------------------------------------------
__global__ void k_gmean(const float* __restrict__ x, bf16* __restrict__ gem) {
  int g = blockIdx.x, c = threadIdx.x;
  float s = 0.f;
  const float* p = x + (size_t)g * NPER * HID + c;
  for (int r = 0; r < NPER; ++r) s += p[(size_t)r * HID];
  gem[g * HID + c] = (bf16)(s * (1.f / NPER));
}

// --- small head final linear (dout <= 16) ----------------------------------
__global__ void k_headfin(const float* __restrict__ h, const float* __restrict__ W,
                          const float* __restrict__ b, float* __restrict__ out,
                          int dout, int ostride, int col0, int zcol) {
  int r = blockIdx.x, t = threadIdx.x;
  if (t < dout) {
    float s = b[t];
    const float* hp = h + (size_t)r * HID;
    for (int k = 0; k < HID; ++k) s += hp[k] * W[k * dout + t];
    out[(size_t)r * ostride + col0 + t] = s;
  }
  if (zcol >= 0 && t == 0) out[(size_t)r * ostride + zcol] = 0.f;
}

// --- per-graph Gram (X X^T / 16), upper-tri packed output -------------------
__global__ __launch_bounds__(256)
void k_scores(const bf16* __restrict__ X, float* __restrict__ out) {
  int g  = blockIdx.y;
  int tp = blockIdx.x;              // 36 tile-pairs (ti<=tj) of 64x64
  int ti = 0;
  while (tp >= (8 - ti)) { tp -= (8 - ti); ++ti; }
  int tj = ti + tp;
  int tid = threadIdx.x;
  int w = tid >> 5, L = tid & 31, half = L >> 4, ln16 = L & 15;
  int iBase = ti * 64 + (w & 3) * 16;
  int jBase = tj * 64 + (w >> 2) * 32;
  const bf16* Xg = X + (size_t)g * NPER * HID;

  FragC acc[2];
#pragma unroll
  for (int nt = 0; nt < 2; ++nt)
#pragma unroll
    for (int e = 0; e < 8; ++e) acc[nt].e[e] = 0.f;

  for (int kt = 0; kt < 8; ++kt) {
    Frag16 a;
    const bf16* ap = Xg + (size_t)(iBase + ln16) * HID + kt * 32 + half * 8;
    a.h[0] = *(const v8bf*)(ap);
    a.h[1] = *(const v8bf*)(ap + 16);
#pragma unroll
    for (int nt = 0; nt < 2; ++nt) {
      const bf16* bp = Xg + (size_t)(jBase + nt * 16 + ln16) * HID + kt * 32 + half * 8;
      Frag16 b;
      b.h[0] = *(const v8bf*)(bp);
      b.h[1] = *(const v8bf*)(bp + 16);
      acc[nt].v = wmma_bf16(a.v, b.v, acc[nt].v);
    }
  }

  float* og = out + (size_t)g * 130816;   // 511*512/2 per graph
#pragma unroll
  for (int nt = 0; nt < 2; ++nt) {
    int j = jBase + nt * 16 + ln16;
#pragma unroll
    for (int v = 0; v < 8; ++v) {
      int i = iBase + v + half * 8;
      if (j > i) {
        int pidx = ((i * (1023 - i)) >> 1) + (j - i - 1);
        og[pidx] = acc[nt].e[v] * 0.0625f;   // 1/sqrt(256)
      }
    }
  }
}

// ---------------------------------------------------------------------------
// Param leaf order (jax tree-flatten: dict keys sorted, lists in order):
//  0 node_class, 1 edge_index, 2 ptr,
//  3..12  add_node_mlp [l1.b,l1.w,l2.b,l2.w,l3.b,l3.w,ln1.b,ln1.g,ln2.b,ln2.g]
//  13..26 convs[0] [gin l1.b,l1.w,l2.b,l2.w,l3.b,l3.w,ln1.b,ln1.g,ln2.b,ln2.g,
//                   post1.b,post1.w,post2.b,post2.w]
//  27..40 convs[1] (same)
//  41..50 edge_class_mlp, 51 embedding, 52..61 exit_mlp, 62..71 node_class_mlp,
//  72 norm.b, 73 norm.g
// ---------------------------------------------------------------------------
extern "C" void kernel_launch(void* const* d_in, const int* in_sizes, int n_in,
                              void* d_out, int out_size, void* d_ws, size_t ws_size,
                              hipStream_t stream) {
  (void)in_sizes; (void)n_in; (void)out_size; (void)ws_size;
  const int* node_class = (const int*)d_in[0];
  const int* edge_index = (const int*)d_in[1];
#define P(i) ((const float*)d_in[(i)])

  // workspace carve (~204 MB)
  char* ws = (char*)d_ws;
  float* xf  = (float*)ws; ws += (size_t)NNODE * HID * 4;
  float* agg = (float*)ws; ws += (size_t)NNODE * HID * 4;
  bf16*  abf = (bf16*)ws;  ws += (size_t)NNODE * HID * 2;
  bf16*  bbf = (bf16*)ws;  ws += (size_t)NNODE * HID * 2;
  bf16*  gem = (bf16*)ws;  ws += (size_t)BGR * HID * 2;
  bf16*  h1  = (bf16*)ws;  ws += (size_t)BGR * HID * 2;
  float* h2  = (float*)ws; ws += (size_t)BGR * HID * 4;
  bf16*  Wt0 = (bf16*)ws;  // 18 * 256*256 bf16
#define WT(s) (Wt0 + (size_t)(s) * HID * HID)

  // bf16-transpose all 256x256 weights used by WMMA GEMMs
  const int widx[18] = {14,16,18,24,26, 28,30,32,38,40, 4,6, 53,55, 63,65, 42,44};
  for (int s = 0; s < 18; ++s)
    k_wtrans<<<HID, HID, 0, stream>>>(P(widx[s]), WT(s));

  // x = embedding[node_class]
  k_embed<<<NNODE, HID, 0, stream>>>(node_class, P(51), xf);

  const float* normb = P(72);
  const float* normg = P(73);

  for (int li = 0; li < 2; ++li) {
    int b = 13 + li * 14;
    int s0 = li * 5;
    // agg = x + scatter_sum(x[src] -> dst)
    k_copy<<<NNODE * HID / 256, 256, 0, stream>>>(xf, agg);
    k_scatter<<<EDGES, HID, 0, stream>>>(edge_index, xf, agg);
    k_tobf<<<NNODE * HID / 256, 256, 0, stream>>>(agg, abf);
    // gin MLP: l1(LN,ReLU) -> l2(LN,ReLU) -> l3
    k_gemm<<<NNODE/64, 256, 0, stream>>>(abf, WT(s0+0), P(b+0), P(b+7), P(b+6), nullptr, nullptr, bbf, NNODE, 3);
    k_gemm<<<NNODE/64, 256, 0, stream>>>(bbf, WT(s0+1), P(b+2), P(b+9), P(b+8), nullptr, nullptr, abf, NNODE, 3);
    k_gemm<<<NNODE/64, 256, 0, stream>>>(abf, WT(s0+2), P(b+4), nullptr, nullptr, nullptr, nullptr, bbf, NNODE, 0);
    // post1(ReLU) -> post2 (+residual if li>0) -> shared LayerNorm
    k_gemm<<<NNODE/64, 256, 0, stream>>>(bbf, WT(s0+3), P(b+10), nullptr, nullptr, nullptr, nullptr, abf, NNODE, 2);
    k_gemm<<<NNODE/64, 256, 0, stream>>>(abf, WT(s0+4), P(b+12), normg, normb,
                                         (li > 0) ? xf : nullptr, xf, bbf, NNODE, 1);
  }

  // graph embeddings (mean over 512 nodes)
  k_gmean<<<BGR, HID, 0, stream>>>(xf, gem);

  float* out = (float*)d_out;
  // heads: add_node -> action[:,0] (zero col 1), exit -> action[:,2],
  //        node_class -> [384..2432), edge_class -> [2432..2944)
  const int hbase[4]   = {3, 52, 62, 41};
  const int hwt[4]     = {10, 12, 14, 16};
  const int hdout[4]   = {1, 1, 16, 4};
  float*    hout[4]    = {out, out, out + 384, out + 2432};
  const int hstride[4] = {3, 3, 16, 4};
  const int hcol0[4]   = {0, 2, 0, 0};
  const int hzcol[4]   = {1, -1, -1, -1};
  for (int hh = 0; hh < 4; ++hh) {
    int b = hbase[hh];
    k_gemm<<<BGR/64, 256, 0, stream>>>(gem, WT(hwt[hh]),   P(b+0), P(b+7), P(b+6), nullptr, nullptr, h1, BGR, 3);
    k_gemm<<<BGR/64, 256, 0, stream>>>(h1,  WT(hwt[hh]+1), P(b+2), P(b+9), P(b+8), nullptr, h2, nullptr, BGR, 3);
    k_headfin<<<BGR, 32, 0, stream>>>(h2, P(b+5), P(b+4), hout[hh],
                                      hdout[hh], hstride[hh], hcol0[hh], hzcol[hh]);
  }

  // edge_index_logits: per-graph upper-tri of X X^T / 16
  k_scores<<<dim3(36, BGR), 256, 0, stream>>>(bbf, out + 2944);
}
